// WaveNetPredictor_33878702031232
// MI455X (gfx1250) — matrix operand order repaired and checked by reading, not compiled, not run on hardware
//
#include <hip/hip_runtime.h>
#include <hip/hip_bf16.h>
#include <math.h>

// ---------------------------------------------------------------------------
// WaveNet on gfx1250: bf16 WMMA (v_wmma_f32_16x16x32_bf16) for all matmuls.
// Epilogues bounce C/D-layout tiles through per-wave LDS so all global
// traffic is packed b128, row-major. Activations use hardware TRANS ops
// (v_exp_f32 / v_rcp_f32) which co-execute with the matrix pipe.
// ---------------------------------------------------------------------------

typedef __attribute__((ext_vector_type(16))) __bf16 v16bf;
typedef __attribute__((ext_vector_type(8)))  float  v8f;

#define DEV __device__ __forceinline__

constexpr int Bn = 16;      // batch
constexpr int Tn = 4096;    // time
constexpr int Dn = 1024;    // input channels
constexpr int Cn = 64;      // residual channels
constexpr int Ln = 40;      // gated layers

DEV v8f wmma_bf16(v16bf a, v16bf b, v8f c) {
  // D = A(16x32 bf16) * B(32x16 bf16) + C(16x16 f32)
  return __builtin_amdgcn_wmma_f32_16x16x32_bf16(
      /*neg_a=*/false, a, /*neg_b=*/false, b,
      /*c_mod=*/(short)0, c, /*reuse_a=*/false, /*reuse_b=*/false);
}

union BF8x { uint4 u; __bf16 h[8]; };

// Fast activations on hardware TRANS ops (v_exp_f32, v_rcp_f32).
DEV float fast_sigmoid(float x) {
  return __builtin_amdgcn_rcpf(1.0f + __expf(-x));
}
DEV float fast_tanh(float x) {
  // tanh(x) = 2*sigmoid(2x) - 1
  return __builtin_fmaf(2.0f, fast_sigmoid(2.0f * x), -1.0f);
}

// A-operand gather (16x32, 16-bit): lane holds row lane&15;
// VGPR 0-3 <- K = kb..kb+7, VGPR 4-7 <- K = kb+16..kb+23, kb = (lane>>4)*8.
// Caller passes p = rowBase + kb (16-byte aligned).
DEV v16bf load_a_bf16(const __bf16* p) {
  BF8x lo, hi;
  lo.u = *reinterpret_cast<const uint4*>(p);
  hi.u = *reinterpret_cast<const uint4*>(p + 16);
  v16bf a;
#pragma unroll
  for (int i = 0; i < 8; ++i) { a[i] = lo.h[i]; a[8 + i] = hi.h[i]; }
  return a;
}

// B-operand gather (32x16, 16-bit): lane holds col lane&15;
// VGPR 0-3 <- K = kb..kb+7, VGPR 4-7 <- K = kb+16..kb+23, kb = (lane>>4)*8.
// Caller passes p = row(cout) base + (lane>>4)*8.
DEV v16bf load_b_bf16(const __bf16* p) {
  BF8x lo, hi;
  lo.u = *reinterpret_cast<const uint4*>(p);
  hi.u = *reinterpret_cast<const uint4*>(p + 16);
  v16bf b;
#pragma unroll
  for (int i = 0; i < 8; ++i) { b[i] = lo.h[i]; b[8 + i] = hi.h[i]; }
  return b;
}

DEV v16bf zero_a() {
  v16bf a;
#pragma unroll
  for (int i = 0; i < 16; ++i) a[i] = (__bf16)0.0f;
  return a;
}

DEV v16bf load_a_f32(const float* p) {           // f32 -> bf16 A tile
  v16bf a;
#pragma unroll
  for (int i = 0; i < 8; ++i) a[i] = (__bf16)p[i];
#pragma unroll
  for (int i = 0; i < 8; ++i) a[8 + i] = (__bf16)p[16 + i];
  return a;
}

DEV v16bf load_a_f32_relu(const float* p) {
  v16bf a;
#pragma unroll
  for (int i = 0; i < 8; ++i) a[i] = (__bf16)fmaxf(p[i], 0.0f);
#pragma unroll
  for (int i = 0; i < 8; ++i) a[8 + i] = (__bf16)fmaxf(p[16 + i], 0.0f);
  return a;
}

DEV v8f splat8(float v) {
  v8f r;
#pragma unroll
  for (int i = 0; i < 8; ++i) r[i] = v;
  return r;
}

// ---------------------------------------------------------------------------
// Weight repack kernels (f32 -> bf16, run each call; cheap, graph-captured).
// ---------------------------------------------------------------------------
__global__ void wnet_cvt(const float* __restrict__ s, __bf16* __restrict__ d, int n) {
  int i = blockIdx.x * blockDim.x + threadIdx.x;
  if (i < n) d[i] = (__bf16)s[i];
}

// (L,C,C,2) tap-last -> (L,2,C,C) tap-major, bf16
__global__ void wnet_repack2(const float* __restrict__ s, __bf16* __restrict__ d, int n) {
  int i = blockIdx.x * blockDim.x + threadIdx.x;
  if (i < n) {
    int ci  = i & 63;
    int co  = (i >> 6) & 63;
    int tap = (i >> 12) & 1;
    int l   = i >> 13;
    d[i] = (__bf16)s[((((l << 6) + co) << 6) + ci) * 2 + tap];
  }
}

__global__ void wnet_zero(float* __restrict__ p, int n) {
  int i = blockIdx.x * blockDim.x + threadIdx.x;
  if (i < n) p[i] = 0.0f;
}

// ---------------------------------------------------------------------------
// Start conv: h = W_start(64x1024) @ x + b   — memory bound on x (268 MB).
// Wave = 16(time) x 64(cout) tile; K-loop over D=1024; packed store via LDS.
// ---------------------------------------------------------------------------
__global__ void __launch_bounds__(256) wnet_start(
    const float* __restrict__ x, const __bf16* __restrict__ w,
    const float* __restrict__ bias, __bf16* __restrict__ h)
{
  __shared__ alignas(16) __bf16 hsh[8][16 * Cn];   // 16 KB

  const int lane  = threadIdx.x & 31;
  const int wv    = threadIdx.x >> 5;
  const int gw    = blockIdx.x * 8 + wv;           // 0..4095
  const int b     = gw >> 8;
  const int mt    = gw & 255;
  const int tbase = mt * 16;
  const int ar    = lane & 15;
  const int kb    = (lane >> 4) * 8;
  const int kh    = (lane >> 4) * 8;

  const float* xrow = x + (size_t)(b * Tn + tbase + ar) * Dn;

  v8f acc[4];
#pragma unroll
  for (int n = 0; n < 4; ++n) acc[n] = splat8(bias[n * 16 + ar]);

  for (int k0 = 0; k0 < Dn; k0 += 32) {
    v16bf A = load_a_f32(xrow + k0 + kb);
#pragma unroll
    for (int n = 0; n < 4; ++n) {
      v16bf Bm = load_b_bf16(w + (size_t)(n * 16 + ar) * Dn + k0 + kh);
      acc[n] = wmma_bf16(A, Bm, acc[n]);
    }
  }

  // C/D-layout scatter into LDS, then packed row-major b128 stores.
  const int mrow = (lane >> 4) * 8;
#pragma unroll
  for (int n = 0; n < 4; ++n)
#pragma unroll
    for (int r = 0; r < 8; ++r)
      hsh[wv][(mrow + r) * Cn + n * 16 + ar] = (__bf16)acc[n][r];

  const int half = (lane >> 4) * 32;               // element offset in row
  const __bf16* zp = &hsh[wv][ar * Cn + half];
  const size_t gbase = (size_t)(b * Tn + tbase + ar) * Cn + half;
#pragma unroll
  for (int q = 0; q < 4; ++q)
    *reinterpret_cast<uint4*>(h + gbase + q * 8) =
        *reinterpret_cast<const uint4*>(zp + q * 8);
}

// ---------------------------------------------------------------------------
// One gated layer. Wave = 16 timesteps x 64 cout. 48 WMMAs per wave.
//   f = tanh(Wd0 h[t-d] + Wd1 h[t] + bd),  g = sigmoid(gate conv)
//   z = f*g ;  skip (+)= Wk z + bk ;  hout = Wr z + br + hin
// z transposed C/D->A through per-wave LDS; res & skip results also bounced
// through LDS so all global stores/RMW are packed b128.
// ---------------------------------------------------------------------------
__global__ void __launch_bounds__(128) wnet_layer(
    const __bf16* __restrict__ hin, __bf16* __restrict__ hout,
    float* __restrict__ skip,
    const __bf16* __restrict__ wd, const __bf16* __restrict__ wg,
    const __bf16* __restrict__ wr, const __bf16* __restrict__ wk,
    const float* __restrict__ bd, const float* __restrict__ bg,
    const float* __restrict__ br, const float* __restrict__ bk,
    int layer, int dil, int first)
{
  __shared__ alignas(16) __bf16 zsh[4][16 * Cn];   // 8 KB  (z, then res)
  __shared__ alignas(16) float  ssh[4][16 * Cn];   // 16 KB (skip f32)

  const int lane  = threadIdx.x & 31;
  const int wv    = threadIdx.x >> 5;
  const int gw    = blockIdx.x * 4 + wv;           // 0..4095
  const int b     = gw >> 8;
  const int mt    = gw & 255;
  const int tbase = mt * 16;
  const int ar    = lane & 15;
  const int kb    = (lane >> 4) * 8;
  const int kh    = (lane >> 4) * 8;
  const int trow  = tbase + ar;

  // ---- A operands: h[t] and causal h[t-d] (zero-padded) ----
  v16bf Ac[2], Ap[2];
  const size_t rowc = (size_t)(b * Tn + trow) * Cn;
#pragma unroll
  for (int kc = 0; kc < 2; ++kc) Ac[kc] = load_a_bf16(hin + rowc + kc * 32 + kb);

  const int tprev = trow - dil;
  if (tprev >= 0) {
    const size_t rowp = (size_t)(b * Tn + tprev) * Cn;
#pragma unroll
    for (int kc = 0; kc < 2; ++kc) Ap[kc] = load_a_bf16(hin + rowp + kc * 32 + kb);
  } else {
    Ap[0] = zero_a(); Ap[1] = zero_a();
  }

  const __bf16* wd0 = wd + (size_t)layer * 2 * Cn * Cn;
  const __bf16* wd1 = wd0 + Cn * Cn;
  const __bf16* wg0 = wg + (size_t)layer * 2 * Cn * Cn;
  const __bf16* wg1 = wg0 + Cn * Cn;
  const __bf16* wrl = wr + (size_t)layer * Cn * Cn;
  const __bf16* wkl = wk + (size_t)layer * Cn * Cn;
  const float*  bdl = bd + layer * Cn;
  const float*  bgl = bg + layer * Cn;
  const float*  brl = br + layer * Cn;
  const float*  bkl = bk + layer * Cn;

  // ---- dilated + gate convs ----
  v8f facc[4], gacc[4];
#pragma unroll
  for (int n = 0; n < 4; ++n) {
    facc[n] = splat8(bdl[n * 16 + ar]);
    gacc[n] = splat8(bgl[n * 16 + ar]);
  }
#pragma unroll
  for (int n = 0; n < 4; ++n) {
#pragma unroll
    for (int kc = 0; kc < 2; ++kc) {
      const size_t bro = (size_t)(n * 16 + ar) * Cn + kc * 32 + kh;
      facc[n] = wmma_bf16(Ap[kc], load_b_bf16(wd0 + bro), facc[n]);
      facc[n] = wmma_bf16(Ac[kc], load_b_bf16(wd1 + bro), facc[n]);
      gacc[n] = wmma_bf16(Ap[kc], load_b_bf16(wg0 + bro), gacc[n]);
      gacc[n] = wmma_bf16(Ac[kc], load_b_bf16(wg1 + bro), gacc[n]);
    }
  }

  // ---- z = tanh(f)*sigmoid(g) -> LDS (C/D layout -> row-major tile) ----
  const int mrow = (lane >> 4) * 8;
#pragma unroll
  for (int n = 0; n < 4; ++n)
#pragma unroll
    for (int r = 0; r < 8; ++r) {
      float zf = fast_tanh(facc[n][r]) * fast_sigmoid(gacc[n][r]);
      zsh[wv][(mrow + r) * Cn + n * 16 + ar] = (__bf16)zf;
    }
  __syncthreads();

  v16bf Z[2];
#pragma unroll
  for (int kc = 0; kc < 2; ++kc)
    Z[kc] = load_a_bf16(&zsh[wv][ar * Cn + kc * 32 + kb]);

  // ---- skip conv -> f32 LDS tile ----
#pragma unroll
  for (int n = 0; n < 4; ++n) {
    v8f s = splat8(bkl[n * 16 + ar]);
#pragma unroll
    for (int kc = 0; kc < 2; ++kc)
      s = wmma_bf16(Z[kc], load_b_bf16(wkl + (size_t)(n * 16 + ar) * Cn + kc * 32 + kh), s);
#pragma unroll
    for (int r = 0; r < 8; ++r)
      ssh[wv][(mrow + r) * Cn + n * 16 + ar] = s[r];
  }

  // ---- residual conv -> bf16 LDS tile (reuse zsh; DS ops are in-order
  //      per wave, and each wave touches only its own tile) ----
#pragma unroll
  for (int n = 0; n < 4; ++n) {
    v8f rr = splat8(brl[n * 16 + ar]);
#pragma unroll
    for (int kc = 0; kc < 2; ++kc)
      rr = wmma_bf16(Z[kc], load_b_bf16(wrl + (size_t)(n * 16 + ar) * Cn + kc * 32 + kh), rr);
#pragma unroll
    for (int r = 0; r < 8; ++r)
      zsh[wv][(mrow + r) * Cn + n * 16 + ar] = (__bf16)rr[r];
  }

  // ---- packed row-major writeback: lane owns 64B of one row ----
  {
    const int half = (lane >> 4) * 32;                       // elements
    const __bf16* zp = &zsh[wv][ar * Cn + half];
    const float*  sp = &ssh[wv][ar * Cn + half];
    const size_t gbase = (size_t)(b * Tn + tbase + ar) * Cn + half;

    // hout = res + h_old  (b128 in / b128 out)
    BF8x rv[4], hv[4], ov[4];
#pragma unroll
    for (int q = 0; q < 4; ++q) rv[q].u = *reinterpret_cast<const uint4*>(zp + q * 8);
#pragma unroll
    for (int q = 0; q < 4; ++q)
      hv[q].u = *reinterpret_cast<const uint4*>(hin + gbase + q * 8);
#pragma unroll
    for (int q = 0; q < 4; ++q) {
#pragma unroll
      for (int e = 0; e < 8; ++e)
        ov[q].h[e] = (__bf16)((float)rv[q].h[e] + (float)hv[q].h[e]);
      *reinterpret_cast<uint4*>(hout + gbase + q * 8) = ov[q].u;
    }

    // skip accumulate (f32, b128)
    if (first) {
#pragma unroll
      for (int q = 0; q < 8; ++q)
        *reinterpret_cast<float4*>(skip + gbase + q * 4) =
            *reinterpret_cast<const float4*>(sp + q * 4);
    } else {
#pragma unroll
      for (int q = 0; q < 8; ++q) {
        float4 a = *reinterpret_cast<const float4*>(sp + q * 4);
        float4 c = *reinterpret_cast<const float4*>(skip + gbase + q * 4);
        a.x += c.x; a.y += c.y; a.z += c.z; a.w += c.w;
        *reinterpret_cast<float4*>(skip + gbase + q * 4) = a;
      }
    }
  }
}

// ---------------------------------------------------------------------------
// End head: e1 = relu(W_end1 @ relu(skip) + b1); partial e2 dot; atomic sums.
// ---------------------------------------------------------------------------
__global__ void __launch_bounds__(128) wnet_end(
    const float* __restrict__ skip, const __bf16* __restrict__ w1,
    const float* __restrict__ b1, const float* __restrict__ w2,
    float* __restrict__ sums)
{
  const int lane  = threadIdx.x & 31;
  const int wv    = threadIdx.x >> 5;
  const int gw    = blockIdx.x * 4 + wv;
  const int b     = gw >> 8;
  const int mt    = gw & 255;
  const int tbase = mt * 16;
  const int ar    = lane & 15;
  const int kb    = (lane >> 4) * 8;
  const int kh    = (lane >> 4) * 8;

  const float* srow = skip + (size_t)(b * Tn + tbase + ar) * Cn;
  v16bf A[2];
#pragma unroll
  for (int kc = 0; kc < 2; ++kc) A[kc] = load_a_f32_relu(srow + kc * 32 + kb);

  float part0 = 0.f, part1 = 0.f, part2 = 0.f, part3 = 0.f;
#pragma unroll
  for (int n = 0; n < 4; ++n) {
    v8f e = splat8(b1[n * 16 + ar]);
#pragma unroll
    for (int kc = 0; kc < 2; ++kc)
      e = wmma_bf16(A[kc], load_b_bf16(w1 + (size_t)(n * 16 + ar) * Cn + kc * 32 + kh), e);
    const int col = n * 16 + ar;
    const float c0 = w2[0 * Cn + col], c1 = w2[1 * Cn + col];
    const float c2 = w2[2 * Cn + col], c3 = w2[3 * Cn + col];
#pragma unroll
    for (int r = 0; r < 8; ++r) {
      float ev = fmaxf(e[r], 0.0f);
      part0 += ev * c0; part1 += ev * c1; part2 += ev * c2; part3 += ev * c3;
    }
  }
  atomicAdd(&sums[b * 4 + 0], part0);
  atomicAdd(&sums[b * 4 + 1], part1);
  atomicAdd(&sums[b * 4 + 2], part2);
  atomicAdd(&sums[b * 4 + 3], part3);
}

__global__ void wnet_final(const float* __restrict__ sums,
                           const float* __restrict__ b2, float* __restrict__ out)
{
  int tid = threadIdx.x;
  if (tid < 64) {
    int j = tid >> 4, b = tid & 15;
    float m = sums[b * 4 + j] / (float)Tn + b2[j];
    float v;
    if (j == 0)      v = m;
    else if (j == 2) v = 1.0f / (1.0f + expf(-m));
    else             v = (m > 20.0f) ? m : log1pf(expf(m));   // softplus
    out[j * 16 + b] = v;
  }
}

// ---------------------------------------------------------------------------
extern "C" void kernel_launch(void* const* d_in, const int* in_sizes, int n_in,
                              void* d_out, int out_size, void* d_ws, size_t ws_size,
                              hipStream_t stream) {
  (void)in_sizes; (void)n_in; (void)out_size; (void)ws_size;
  const float* x       = (const float*)d_in[0];
  const float* w_start = (const float*)d_in[1];
  const float* b_start = (const float*)d_in[2];
  const float* w_dil   = (const float*)d_in[3];
  const float* b_dil   = (const float*)d_in[4];
  const float* w_gate  = (const float*)d_in[5];
  const float* b_gate  = (const float*)d_in[6];
  const float* w_res   = (const float*)d_in[7];
  const float* b_res   = (const float*)d_in[8];
  const float* w_skip  = (const float*)d_in[9];
  const float* b_skip  = (const float*)d_in[10];
  const float* w_end1  = (const float*)d_in[11];
  const float* b_end1  = (const float*)d_in[12];
  const float* w_end2  = (const float*)d_in[13];
  const float* b_end2  = (const float*)d_in[14];
  float* out = (float*)d_out;

  // Workspace carve-out (256B aligned)
  char* ws = (char*)d_ws;
  size_t off = 0;
  auto carve = [&](size_t bytes) -> char* {
    char* p = ws + off;
    off = (off + bytes + 255) & ~size_t(255);
    return p;
  };
  __bf16* hA       = (__bf16*)carve((size_t)Bn * Tn * Cn * 2);
  __bf16* hB       = (__bf16*)carve((size_t)Bn * Tn * Cn * 2);
  float*  skip     = (float*) carve((size_t)Bn * Tn * Cn * 4);
  __bf16* wstart_b = (__bf16*)carve((size_t)Cn * Dn * 2);
  __bf16* wdil_b   = (__bf16*)carve((size_t)Ln * 2 * Cn * Cn * 2);
  __bf16* wgate_b  = (__bf16*)carve((size_t)Ln * 2 * Cn * Cn * 2);
  __bf16* wres_b   = (__bf16*)carve((size_t)Ln * Cn * Cn * 2);
  __bf16* wskip_b  = (__bf16*)carve((size_t)Ln * Cn * Cn * 2);
  __bf16* wend1_b  = (__bf16*)carve((size_t)Cn * Cn * 2);
  float*  sums     = (float*) carve((size_t)Bn * 4 * 4);

  // Weight repack (f32 -> bf16; dil/gate to tap-major)
  {
    int n;
    n = Cn * Dn;            wnet_cvt    <<<(n + 255) / 256, 256, 0, stream>>>(w_start, wstart_b, n);
    n = Ln * 2 * Cn * Cn;   wnet_repack2<<<(n + 255) / 256, 256, 0, stream>>>(w_dil,   wdil_b,   n);
    n = Ln * 2 * Cn * Cn;   wnet_repack2<<<(n + 255) / 256, 256, 0, stream>>>(w_gate,  wgate_b,  n);
    n = Ln * Cn * Cn;       wnet_cvt    <<<(n + 255) / 256, 256, 0, stream>>>(w_res,   wres_b,   n);
    n = Ln * Cn * Cn;       wnet_cvt    <<<(n + 255) / 256, 256, 0, stream>>>(w_skip,  wskip_b,  n);
    n = Cn * Cn;            wnet_cvt    <<<(n + 255) / 256, 256, 0, stream>>>(w_end1,  wend1_b,  n);
  }
  wnet_zero<<<1, 64, 0, stream>>>(sums, Bn * 4);

  // Start conv: 4096 waves, 8 waves/block
  wnet_start<<<512, 256, 0, stream>>>(x, wstart_b, b_start, hA);

  // 40 gated layers, double-buffered h
  __bf16* hin = hA;
  __bf16* hout = hB;
  for (int idx = 0; idx < Ln; ++idx) {
    int d = 1 << (idx % 10);
    wnet_layer<<<1024, 128, 0, stream>>>(hin, hout, skip,
                                         wdil_b, wgate_b, wres_b, wskip_b,
                                         b_dil, b_gate, b_res, b_skip,
                                         idx, d, idx == 0 ? 1 : 0);
    __bf16* t = hin; hin = hout; hout = t;
  }

  // End head + finalize
  wnet_end<<<1024, 128, 0, stream>>>(skip, wend1_b, b_end1, w_end2, sums);
  wnet_final<<<1, 64, 0, stream>>>(sums, b_end2, out);
}
